// SparseUNet_43722767073325
// MI455X (gfx1250) — compile-verified, matrix operand order checked by reading, not run on hardware
//
#include <hip/hip_runtime.h>
#include <hip/hip_bf16.h>
#include <stdint.h>

// ---------------------------------------------------------------------------
// Sparse U-Net (MinkowskiEngine-style masked dense emulation) for MI455X
// (gfx1250). All channel GEMMs (3x3 convs via im2col-in-LDS, 2x2 stride-2
// transpose convs as 4 shared-A 1x1 GEMMs) run on v_wmma_f32_16x16x32_f16
// with f16 activations and f32 accumulation. Conv input tiles are staged to
// LDS with gfx1250 async global->LDS loads (ASYNCcnt). Masking/pooling/
// 1-channel edge convs stay on VALU (negligible FLOPs).
// ---------------------------------------------------------------------------

typedef __attribute__((ext_vector_type(16))) _Float16 v16h;
typedef __attribute__((ext_vector_type(8)))  float    v8f;

// K index inside a 16x32 f16 A-fragment (ISA 7.12.2): half h = lane>>4,
// VGPR v (2 f16 each), p = lo/hi half.
__device__ __forceinline__ int kmapf(int h, int v, int p) {
  return (v < 4) ? (h * 8 + v * 2 + p) : (16 + h * 8 + (v - 4) * 2 + p);
}

union AFrag { v16h h; uint32_t u[8]; };

// ---------------------------------------------------------------------------
// Weight packing: f32 [taps, Ci, Co] -> f16 B-fragment stream.
// Layout: (((t*KC + kc)*(Co/16) + ct)*32 + lane)*16 + j, where lane's column
// is N = ct*16 + (lane&15) and j enumerates the lane's 16 K values via kmapf.
// Zero-fills ci >= Ci (K padding for Ci=16 -> CIT=32).
// ---------------------------------------------------------------------------
__global__ void k_pack_w(const float* __restrict__ w, _Float16* __restrict__ out,
                         int taps, int Ci, int KC, int Co) {
  int total = taps * KC * (Co / 16) * 512;
  int idx = blockIdx.x * blockDim.x + threadIdx.x;
  if (idx >= total) return;
  int j    = idx & 15;
  int lane = (idx >> 4) & 31;
  int rest = idx >> 9;
  int COT  = Co / 16;
  int ct = rest % COT; rest /= COT;
  int kc = rest % KC;
  int t  = rest / KC;
  int h  = lane >> 4;
  int K  = kmapf(h, j >> 1, j & 1);
  int ci = kc * 32 + K;
  int co = ct * 16 + (lane & 15);
  float v = (ci < Ci) ? w[((size_t)t * Ci + ci) * Co + co] : 0.0f;
  out[idx] = (_Float16)v;
}

// ---------------------------------------------------------------------------
// Masked 3x3 conv (SAME), f16 in/out, WMMA core.
// inA/inB are the two concat halves (inB may be null, CIB==0). Inputs are
// already zero at unoccupied sites, so the conv is dense; output is zeroed
// where mOut==0. Workgroup: 128 threads = 4 waves; tile = 16x16 output pixels,
// each wave owns 4 rows (M = 16 pixels along x), loops all Co tiles.
// Halo tile staged via global_load_async_to_lds_b128 (ASYNCcnt); zero-fill
// for out-of-image halo / channel padding uses plain LDS stores.
// ---------------------------------------------------------------------------
template<int CIA, int CIB, int CO>
__global__ void __launch_bounds__(128)
k_conv3(const _Float16* __restrict__ inA,
        const _Float16* __restrict__ inB,
        const unsigned char* __restrict__ mOut,
        const _Float16* __restrict__ wpack,
        _Float16* __restrict__ out,
        int H, int W) {
  constexpr int CIS = CIA + CIB;
  constexpr int CIT = (CIS < 32) ? 32 : CIS;   // K padded to multiple of 32
  constexpr int KC  = CIT / 32;
  constexpr int COT = CO / 16;
  extern __shared__ unsigned char smem[];      // 324*CIT f16 + 256 mask bytes

  const int tilesX = W >> 4;
  const int tX  = blockIdx.x % tilesX;
  const int tY  = blockIdx.x / tilesX;
  const int b   = blockIdx.y;
  const int tid = threadIdx.x;

  // ---- stage 18x18 halo tile (zero-padded at borders and channel pad) ----
  {
    constexpr int CG = CIT / 8;                // 16B chunks per pixel
    constexpr int nChunks = 324 * CG;
    // Generic LDS addresses carry the LDS byte offset in bits [31:0]
    // (ISA 10.2 aperture rules), which is what the async-load VDST wants.
    const uint32_t ldsBase = (uint32_t)(uintptr_t)(void*)smem;
    for (int q = tid; q < nChunks; q += 128) {
      int p  = q / CG;
      int c0 = (q % CG) * 8;
      int ly = p / 18, lx = p % 18;
      int gy = tY * 16 + ly - 1;
      int gx = tX * 16 + lx - 1;
      const _Float16* src = nullptr;
      if (gy >= 0 && gy < H && gx >= 0 && gx < W) {
        size_t pix = (size_t)(b * H + gy) * W + gx;
        if (c0 < CIA)                      src = inA + pix * CIA + c0;
        else if (CIB > 0 && c0 < CIS)      src = inB + pix * CIB + (c0 - CIA);
      }
      if (src) {
        uint32_t dst = ldsBase + (uint32_t)(p * CIT + c0) * 2;
        asm volatile("global_load_async_to_lds_b128 %0, %1, off"
                     :: "v"(dst), "v"((unsigned long long)(uintptr_t)src)
                     : "memory");
      } else {
        *(float4*)(smem + (size_t)(p * CIT + c0) * 2) =
            make_float4(0.f, 0.f, 0.f, 0.f);
      }
    }
    unsigned char* smk = smem + (size_t)324 * CIT * 2;
    for (int i = tid; i < 256; i += 128) {
      int ly = i >> 4, lx = i & 15;
      smk[i] = mOut[(size_t)(b * H + tY * 16 + ly) * W + (tX * 16 + lx)];
    }
    asm volatile("s_wait_asynccnt 0" ::: "memory");
  }
  __syncthreads();

  const uint32_t* lds32 = (const uint32_t*)smem;
  const unsigned char* smk = smem + (size_t)324 * CIT * 2;
  const int wv   = tid >> 5;
  const int lane = tid & 31;
  const int h    = lane >> 4;
  const int nn   = lane & 15;     // N column (output channel within tile)
  const int mrow = lane & 15;     // A-matrix row (pixel x within tile)

  for (int ct = 0; ct < COT; ++ct) {
    for (int rr = 0; rr < 4; ++rr) {
      const int row = wv * 4 + rr;
      v8f acc = {};
      for (int t = 0; t < 9; ++t) {
        const int ty = t / 3, tx = t % 3;
        const int pixbase = ((row + ty) * 18 + (mrow + tx)) * CIT;
        #pragma unroll
        for (int kc = 0; kc < KC; ++kc) {
          AFrag a;
          #pragma unroll
          for (int v = 0; v < 8; ++v)
            a.u[v] = lds32[(pixbase + kc * 32 + kmapf(h, v, 0)) >> 1];
          const _Float16* bp =
              wpack + ((size_t)((t * KC + kc) * COT + ct) << 9) + lane * 16;
          __builtin_prefetch(bp + 512, 0, 1);
          v16h bf = *(const v16h*)bp;
          acc = __builtin_amdgcn_wmma_f32_16x16x32_f16(
              false, a.h, false, bf, (short)0, acc, false, false);
        }
      }
      const int y = tY * 16 + row;
      #pragma unroll
      for (int i = 0; i < 8; ++i) {
        const int M = h * 8 + i;               // C/D layout: M split at 8
        const int x = tX * 16 + M;
        float vv = smk[row * 16 + M] ? acc[i] : 0.0f;
        out[((size_t)(b * H + y) * W + x) * CO + ct * 16 + nn] = (_Float16)vv;
      }
    }
  }
}

// ---------------------------------------------------------------------------
// Kernel-2 stride-2 transpose conv ("generative"): 4 taps = 4 independent
// 1x1 GEMMs sharing each A-fragment. M = 16 flattened input pixels per wave.
// Output masked by the input-level mask replicated 2x2.
// ---------------------------------------------------------------------------
template<int CI, int CO>
__global__ void __launch_bounds__(128)
k_tconv2(const _Float16* __restrict__ x,
         const unsigned char* __restrict__ m,
         const _Float16* __restrict__ wpack,
         _Float16* __restrict__ out,
         int Hin, int Win) {
  constexpr int KC  = CI / 32;
  constexpr int COT = CO / 16;
  const int lane = threadIdx.x & 31;
  const int wv   = threadIdx.x >> 5;
  const int h    = lane >> 4;
  const int nn   = lane & 15;
  const int pixBase = blockIdx.x * 64 + wv * 16;
  const int ct = blockIdx.y;

  v8f acc0 = {}, acc1 = {}, acc2 = {}, acc3 = {};
  const size_t abase = (size_t)(pixBase + (lane & 15)) * CI;
  const size_t tapStride = (size_t)KC * COT * 512;
  #pragma unroll
  for (int kc = 0; kc < KC; ++kc) {
    AFrag a;
    #pragma unroll
    for (int v = 0; v < 8; ++v)
      a.u[v] = *(const uint32_t*)(x + abase + kc * 32 + kmapf(h, v, 0));
    const _Float16* wb = wpack + ((size_t)kc * COT + ct) * 512 + lane * 16;
    v16h b0 = *(const v16h*)(wb);
    v16h b1 = *(const v16h*)(wb + tapStride);
    v16h b2 = *(const v16h*)(wb + 2 * tapStride);
    v16h b3 = *(const v16h*)(wb + 3 * tapStride);
    acc0 = __builtin_amdgcn_wmma_f32_16x16x32_f16(false, a.h, false, b0, (short)0, acc0, false, false);
    acc1 = __builtin_amdgcn_wmma_f32_16x16x32_f16(false, a.h, false, b1, (short)0, acc1, false, false);
    acc2 = __builtin_amdgcn_wmma_f32_16x16x32_f16(false, a.h, false, b2, (short)0, acc2, false, false);
    acc3 = __builtin_amdgcn_wmma_f32_16x16x32_f16(false, a.h, false, b3, (short)0, acc3, false, false);
  }

  const int Hout = Hin * 2, Wout = Win * 2;
  #pragma unroll
  for (int i = 0; i < 8; ++i) {
    const int M = h * 8 + i;
    const int p = pixBase + M;
    const int b_  = p / (Hin * Win);
    const int rem = p % (Hin * Win);
    const int hy = rem / Win, wx = rem % Win;
    const bool mok = m[p] != 0;
    float v0 = mok ? acc0[i] : 0.f;
    float v1 = mok ? acc1[i] : 0.f;
    float v2 = mok ? acc2[i] : 0.f;
    float v3 = mok ? acc3[i] : 0.f;
    size_t rowA = (size_t)(b_ * Hout + 2 * hy) * Wout + 2 * wx;
    size_t rowB = (size_t)(b_ * Hout + 2 * hy + 1) * Wout + 2 * wx;
    out[(rowA    ) * CO + ct * 16 + nn] = (_Float16)v0;   // (di=0,dj=0)
    out[(rowA + 1) * CO + ct * 16 + nn] = (_Float16)v1;   // (0,1)
    out[(rowB    ) * CO + ct * 16 + nn] = (_Float16)v2;   // (1,0)
    out[(rowB + 1) * CO + ct * 16 + nn] = (_Float16)v3;   // (1,1)
  }
}

// ---------------------------------------------------------------------------
// First conv: Ci=1 -> 16, masked input AND output (raw x is nonzero off-mask).
// Also emits the uint8 level-0 mask.
// ---------------------------------------------------------------------------
__global__ void k_conv1(const float* __restrict__ x, const int* __restrict__ mask,
                        const float* __restrict__ w, _Float16* __restrict__ c1,
                        unsigned char* __restrict__ m0, int Bn, int H, int W) {
  int idx = blockIdx.x * blockDim.x + threadIdx.x;
  int total = Bn * H * W;
  if (idx >= total) return;
  int b   = idx / (H * W);
  int rem = idx % (H * W);
  int y = rem / W, xx = rem % W;
  float acc[16];
  #pragma unroll
  for (int c = 0; c < 16; ++c) acc[c] = 0.f;
  for (int t = 0; t < 9; ++t) {
    int ny = y + t / 3 - 1, nx = xx + t % 3 - 1;
    if (ny < 0 || ny >= H || nx < 0 || nx >= W) continue;
    size_t n = (size_t)(b * H + ny) * W + nx;
    if (!mask[n]) continue;
    float xv = x[n];
    #pragma unroll
    for (int c = 0; c < 16; ++c) acc[c] += xv * w[t * 16 + c];
  }
  bool mm = mask[idx] != 0;
  m0[idx] = mm ? 1 : 0;
  #pragma unroll
  for (int c = 0; c < 16; ++c)
    c1[(size_t)idx * 16 + c] = (_Float16)(mm ? acc[c] : 0.f);
}

// Masked 2x2 stride-2 max pool + mask reduction.
__global__ void k_pool(const _Float16* __restrict__ x, const unsigned char* __restrict__ m,
                       _Float16* __restrict__ y, unsigned char* __restrict__ mo,
                       int Bn, int Ho, int Wo, int C) {
  int idx = blockIdx.x * blockDim.x + threadIdx.x;
  int total = Bn * Ho * Wo * C;
  if (idx >= total) return;
  int c  = idx % C;
  int op = idx / C;
  int b  = op / (Ho * Wo);
  int r  = op % (Ho * Wo);
  int oy = r / Wo, ox = r % Wo;
  int Hi = Ho * 2, Wi = Wo * 2;
  float best = -3.0e38f;
  int any = 0;
  #pragma unroll
  for (int d = 0; d < 4; ++d) {
    int iy = oy * 2 + (d >> 1), ix = ox * 2 + (d & 1);
    size_t pi = (size_t)(b * Hi + iy) * Wi + ix;
    int mv = m[pi];
    any |= mv;
    float xv = mv ? (float)x[pi * C + c] : -3.0e38f;
    best = best > xv ? best : xv;
  }
  y[idx] = (_Float16)(any ? best : 0.f);
  if (c == 0) mo[op] = any ? 1 : 0;
}

// mc = repeat(m_coarse, 2x2) | m_fine
__global__ void k_upmask(const unsigned char* __restrict__ mcoarse,
                         const unsigned char* __restrict__ mfine,
                         unsigned char* __restrict__ dst, int Bn, int Hf, int Wf) {
  int idx = blockIdx.x * blockDim.x + threadIdx.x;
  int total = Bn * Hf * Wf;
  if (idx >= total) return;
  int b = idx / (Hf * Wf);
  int r = idx % (Hf * Wf);
  int y = r / Wf, x = r % Wf;
  unsigned char cc =
      mcoarse[(size_t)(b * (Hf / 2) + (y >> 1)) * (Wf / 2) + (x >> 1)];
  dst[idx] = (cc | mfine[idx]) ? 1 : 0;
}

// Final masked 1x1 conv 16 -> 3, f32 output.
__global__ void k_out1(const _Float16* __restrict__ r1, const unsigned char* __restrict__ mc,
                       const float* __restrict__ w, float* __restrict__ out,
                       int Bn, int H, int W) {
  int idx = blockIdx.x * blockDim.x + threadIdx.x;
  int total = Bn * H * W;
  if (idx >= total) return;
  float a0 = 0.f, a1 = 0.f, a2 = 0.f;
  if (mc[idx]) {
    #pragma unroll
    for (int ci = 0; ci < 16; ++ci) {
      float xv = (float)r1[(size_t)idx * 16 + ci];
      a0 += xv * w[ci * 3 + 0];
      a1 += xv * w[ci * 3 + 1];
      a2 += xv * w[ci * 3 + 2];
    }
  }
  out[(size_t)idx * 3 + 0] = a0;
  out[(size_t)idx * 3 + 1] = a1;
  out[(size_t)idx * 3 + 2] = a2;
}

__global__ void k_f16tof32(const _Float16* __restrict__ src, float* __restrict__ dst, int n) {
  int idx = blockIdx.x * blockDim.x + threadIdx.x;
  if (idx < n) dst[idx] = (float)src[idx];
}

// ---------------------------------------------------------------------------
// Host orchestration
// ---------------------------------------------------------------------------
extern "C" void kernel_launch(void* const* d_in, const int* in_sizes, int n_in,
                              void* d_out, int out_size, void* d_ws, size_t ws_size,
                              hipStream_t stream) {
  (void)in_sizes; (void)n_in; (void)out_size; (void)ws_size;
  const int Bn = 2, H0 = 1024, W0 = 1024;

  const float* x       = (const float*)d_in[0];
  const int*   mask    = (const int*)d_in[1];
  const float* w_ds1   = (const float*)d_in[2];
  const float* w_ds2   = (const float*)d_in[3];
  const float* w_ds3   = (const float*)d_in[4];
  const float* w_ds4   = (const float*)d_in[5];
  const float* w_brid  = (const float*)d_in[6];
  const float* wt4     = (const float*)d_in[7];
  const float* w_us4   = (const float*)d_in[8];
  const float* wt3     = (const float*)d_in[9];
  const float* w_us3   = (const float*)d_in[10];
  const float* wt2     = (const float*)d_in[11];
  const float* w_us2   = (const float*)d_in[12];
  const float* wt1     = (const float*)d_in[13];
  const float* w_us1   = (const float*)d_in[14];
  const float* w_out   = (const float*)d_in[15];

  char* ws = (char*)d_ws;
  size_t off = 0;
  auto alloc = [&](size_t bytes) -> char* {
    char* p = ws + off;
    off += (bytes + 255) & ~(size_t)255;
    return p;
  };

  // activation / mask buffers (f16 activations; u8 masks)
  unsigned char* m0  = (unsigned char*)alloc((size_t)Bn * H0 * W0);
  _Float16* c1  = (_Float16*)alloc((size_t)Bn * 1024 * 1024 * 16 * 2);
  _Float16* p1  = (_Float16*)alloc((size_t)Bn * 512 * 512 * 16 * 2);
  unsigned char* m1  = (unsigned char*)alloc((size_t)Bn * 512 * 512);
  _Float16* c2  = (_Float16*)alloc((size_t)Bn * 512 * 512 * 32 * 2);
  _Float16* p2  = (_Float16*)alloc((size_t)Bn * 256 * 256 * 32 * 2);
  unsigned char* m2  = (unsigned char*)alloc((size_t)Bn * 256 * 256);
  _Float16* c3  = (_Float16*)alloc((size_t)Bn * 256 * 256 * 64 * 2);
  _Float16* p3  = (_Float16*)alloc((size_t)Bn * 128 * 128 * 64 * 2);
  unsigned char* m3  = (unsigned char*)alloc((size_t)Bn * 128 * 128);
  _Float16* c4  = (_Float16*)alloc((size_t)Bn * 128 * 128 * 128 * 2);
  _Float16* p4  = (_Float16*)alloc((size_t)Bn * 64 * 64 * 128 * 2);
  unsigned char* m4  = (unsigned char*)alloc((size_t)Bn * 64 * 64);
  _Float16* bv  = (_Float16*)alloc((size_t)Bn * 64 * 64 * 256 * 2);
  _Float16* u4  = (_Float16*)alloc((size_t)Bn * 128 * 128 * 128 * 2);
  unsigned char* mc4 = (unsigned char*)alloc((size_t)Bn * 128 * 128);
  _Float16* r4  = (_Float16*)alloc((size_t)Bn * 128 * 128 * 128 * 2);
  _Float16* u3  = (_Float16*)alloc((size_t)Bn * 256 * 256 * 64 * 2);
  unsigned char* mc3 = (unsigned char*)alloc((size_t)Bn * 256 * 256);
  _Float16* r3  = (_Float16*)alloc((size_t)Bn * 256 * 256 * 64 * 2);
  _Float16* u2  = (_Float16*)alloc((size_t)Bn * 512 * 512 * 32 * 2);
  unsigned char* mc2 = (unsigned char*)alloc((size_t)Bn * 512 * 512);
  _Float16* r2  = (_Float16*)alloc((size_t)Bn * 512 * 512 * 32 * 2);
  _Float16* u1  = (_Float16*)alloc((size_t)Bn * 1024 * 1024 * 16 * 2);
  unsigned char* mc1 = (unsigned char*)alloc((size_t)Bn * 1024 * 1024);
  _Float16* r1  = (_Float16*)alloc((size_t)Bn * 1024 * 1024 * 16 * 2);

  // packed weights (f16 B-fragment streams)
  _Float16* pw_ds2 = (_Float16*)alloc((size_t)9 * 1 * 2  * 512 * 2);
  _Float16* pw_ds3 = (_Float16*)alloc((size_t)9 * 1 * 4  * 512 * 2);
  _Float16* pw_ds4 = (_Float16*)alloc((size_t)9 * 2 * 8  * 512 * 2);
  _Float16* pw_br  = (_Float16*)alloc((size_t)9 * 4 * 16 * 512 * 2);
  _Float16* pw_us4 = (_Float16*)alloc((size_t)9 * 8 * 8  * 512 * 2);
  _Float16* pw_us3 = (_Float16*)alloc((size_t)9 * 4 * 4  * 512 * 2);
  _Float16* pw_us2 = (_Float16*)alloc((size_t)9 * 2 * 2  * 512 * 2);
  _Float16* pw_us1 = (_Float16*)alloc((size_t)9 * 1 * 1  * 512 * 2);
  _Float16* pt4    = (_Float16*)alloc((size_t)4 * 8 * 8  * 512 * 2);
  _Float16* pt3    = (_Float16*)alloc((size_t)4 * 4 * 4  * 512 * 2);
  _Float16* pt2    = (_Float16*)alloc((size_t)4 * 2 * 2  * 512 * 2);
  _Float16* pt1    = (_Float16*)alloc((size_t)4 * 1 * 1  * 512 * 2);

  auto pack = [&](const float* w, _Float16* dst, int taps, int Ci, int KC, int Co) {
    int total = taps * KC * (Co / 16) * 512;
    k_pack_w<<<(total + 255) / 256, 256, 0, stream>>>(w, dst, taps, Ci, KC, Co);
  };
  pack(w_ds2, pw_ds2, 9, 16,  1, 32);
  pack(w_ds3, pw_ds3, 9, 32,  1, 64);
  pack(w_ds4, pw_ds4, 9, 64,  2, 128);
  pack(w_brid, pw_br, 9, 128, 4, 256);
  pack(w_us4, pw_us4, 9, 256, 8, 128);
  pack(w_us3, pw_us3, 9, 128, 4, 64);
  pack(w_us2, pw_us2, 9, 64,  2, 32);
  pack(w_us1, pw_us1, 9, 32,  1, 16);
  pack(wt4,   pt4,    4, 256, 8, 128);
  pack(wt3,   pt3,    4, 128, 4, 64);
  pack(wt2,   pt2,    4, 64,  2, 32);
  pack(wt1,   pt1,    4, 32,  1, 16);

  auto pool = [&](const _Float16* xi, const unsigned char* mi, _Float16* yo,
                  unsigned char* mo2, int Ho, int Wo, int C) {
    int total = Bn * Ho * Wo * C;
    k_pool<<<(total + 255) / 256, 256, 0, stream>>>(xi, mi, yo, mo2, Bn, Ho, Wo, C);
  };
  auto upmask = [&](const unsigned char* mc_, const unsigned char* mf,
                    unsigned char* dst, int Hf, int Wf) {
    int total = Bn * Hf * Wf;
    k_upmask<<<(total + 255) / 256, 256, 0, stream>>>(mc_, mf, dst, Bn, Hf, Wf);
  };

  // ---- encoder ----
  {
    int total = Bn * H0 * W0;
    k_conv1<<<(total + 255) / 256, 256, 0, stream>>>(x, mask, w_ds1, c1, m0, Bn, H0, W0);
  }
  pool(c1, m0, p1, m1, 512, 512, 16);
  {
    dim3 g((512 / 16) * (512 / 16), Bn);
    k_conv3<16, 0, 32><<<g, 128, 324 * 32 * 2 + 256, stream>>>(p1, nullptr, m1, pw_ds2, c2, 512, 512);
  }
  pool(c2, m1, p2, m2, 256, 256, 32);
  {
    dim3 g((256 / 16) * (256 / 16), Bn);
    k_conv3<32, 0, 64><<<g, 128, 324 * 32 * 2 + 256, stream>>>(p2, nullptr, m2, pw_ds3, c3, 256, 256);
  }
  pool(c3, m2, p3, m3, 128, 128, 64);
  {
    dim3 g((128 / 16) * (128 / 16), Bn);
    k_conv3<64, 0, 128><<<g, 128, 324 * 64 * 2 + 256, stream>>>(p3, nullptr, m3, pw_ds4, c4, 128, 128);
  }
  pool(c4, m3, p4, m4, 64, 64, 128);
  {
    dim3 g((64 / 16) * (64 / 16), Bn);
    k_conv3<128, 0, 256><<<g, 128, 324 * 128 * 2 + 256, stream>>>(p4, nullptr, m4, pw_br, bv, 64, 64);
  }

  // ---- decoder ----
  { dim3 g(Bn * 64 * 64 / 64, 128 / 16);
    k_tconv2<256, 128><<<g, 128, 0, stream>>>(bv, m4, pt4, u4, 64, 64); }
  upmask(m4, m3, mc4, 128, 128);
  { dim3 g((128 / 16) * (128 / 16), Bn);
    k_conv3<128, 128, 128><<<g, 128, 324 * 256 * 2 + 256, stream>>>(u4, c4, mc4, pw_us4, r4, 128, 128); }

  { dim3 g(Bn * 128 * 128 / 64, 64 / 16);
    k_tconv2<128, 64><<<g, 128, 0, stream>>>(r4, mc4, pt3, u3, 128, 128); }
  upmask(mc4, m2, mc3, 256, 256);
  { dim3 g((256 / 16) * (256 / 16), Bn);
    k_conv3<64, 64, 64><<<g, 128, 324 * 128 * 2 + 256, stream>>>(u3, c3, mc3, pw_us3, r3, 256, 256); }

  { dim3 g(Bn * 256 * 256 / 64, 32 / 16);
    k_tconv2<64, 32><<<g, 128, 0, stream>>>(r3, mc3, pt2, u2, 256, 256); }
  upmask(mc3, m1, mc2, 512, 512);
  { dim3 g((512 / 16) * (512 / 16), Bn);
    k_conv3<32, 32, 32><<<g, 128, 324 * 64 * 2 + 256, stream>>>(u2, c2, mc2, pw_us2, r2, 512, 512); }

  { dim3 g(Bn * 512 * 512 / 64, 16 / 16);
    k_tconv2<32, 16><<<g, 128, 0, stream>>>(r2, mc2, pt1, u1, 512, 512); }
  upmask(mc2, m0, mc1, 1024, 1024);
  { dim3 g((1024 / 16) * (1024 / 16), Bn);
    k_conv3<16, 16, 16><<<g, 128, 324 * 32 * 2 + 256, stream>>>(u1, c1, mc1, pw_us1, r1, 1024, 1024); }

  // ---- outputs: (out f32 [B,H,W,3]) then (c2 f32 [B,512,512,32]) ----
  {
    int total = Bn * H0 * W0;
    k_out1<<<(total + 255) / 256, 256, 0, stream>>>(r1, mc1, w_out, (float*)d_out, Bn, H0, W0);
  }
  {
    int nc2 = Bn * 512 * 512 * 32;
    float* c2out = (float*)d_out + (size_t)Bn * H0 * W0 * 3;
    k_f16tof32<<<(nc2 + 255) / 256, 256, 0, stream>>>(c2, c2out, nc2);
  }
}
